// LSTM_small2_90434831384801
// MI455X (gfx1250) — matrix operand order repaired and checked by reading, not compiled, not run on hardware
//
#include <hip/hip_runtime.h>
#include <hip/hip_bf16.h>

typedef unsigned short ushort_t;
typedef __attribute__((ext_vector_type(16))) __bf16 v16bf;
typedef __attribute__((ext_vector_type(8)))  float  v8f;

#define LDS_STRIDE 40   // padded bf16 stride for 32-wide K tiles (keeps 16B align, breaks bank pathologies)

#if defined(__has_builtin)
#if __has_builtin(__builtin_amdgcn_global_load_async_to_lds_b128)
#define HAS_ASYNC_LDS 1
#endif
#endif
#ifndef HAS_ASYNC_LDS
#define HAS_ASYNC_LDS 0
#endif

#if HAS_ASYNC_LDS
typedef int v4i_t __attribute__((vector_size(16)));
typedef __attribute__((address_space(1))) v4i_t as1_v4i;
typedef __attribute__((address_space(3))) v4i_t as3_v4i;
#if __has_builtin(__builtin_amdgcn_s_wait_asynccnt)
#define WAIT_ASYNC(n) __builtin_amdgcn_s_wait_asynccnt(n)
#else
#define WAIT_ASYNC(n) asm volatile("s_wait_asynccnt %0" :: "n"(n) : "memory")
#endif
#else
#define WAIT_ASYNC(n) ((void)0)
#endif

// ---------- helpers ----------
__device__ __forceinline__ ushort_t f2bf(float f) {
    unsigned u = __float_as_uint(f);
    unsigned r = u + 0x7FFFu + ((u >> 16) & 1u);   // round-to-nearest-even
    return (ushort_t)(r >> 16);
}
__device__ __forceinline__ float sigm(float x) { return 1.0f / (1.0f + __expf(-x)); }

// ---------- weight / input packing ----------
__global__ void pack_lstm_w(const float* __restrict__ Wih, const float* __restrict__ Whh,
                            ushort_t* __restrict__ Wp) {
    // 2 layers x 6144 x 3072 ; k<1536 -> Wih, else Whh
    long idx = (long)blockIdx.x * 256 + threadIdx.x;
    const long total = 2L * 6144 * 3072;
    if (idx >= total) return;
    int k2 = (int)(idx % 3072);
    long nl = idx / 3072;
    int n = (int)(nl % 6144);
    int l = (int)(nl / 6144);
    float v = (k2 < 1536) ? Wih[((long)l * 6144 + n) * 1536 + k2]
                          : Whh[((long)l * 6144 + n) * 1536 + (k2 - 1536)];
    Wp[idx] = f2bf(v);
}

__global__ void pack_bias(const float* __restrict__ bih, const float* __restrict__ bhh,
                          float* __restrict__ bp, int n) {
    int i = blockIdx.x * 256 + threadIdx.x;
    if (i < n) bp[i] = bih[i] + bhh[i];
}

__global__ void transpose_w(const float* __restrict__ W /*K x N*/, ushort_t* __restrict__ Wt /*N x K*/,
                            int K, int N) {
    long idx = (long)blockIdx.x * 256 + threadIdx.x;
    if (idx >= (long)K * N) return;
    int k = (int)(idx % K);
    int n = (int)(idx / K);
    Wt[idx] = f2bf(W[(long)k * N + n]);
}

__global__ void build_seq(const float* __restrict__ x, ushort_t* __restrict__ seq) {
    // seq[t][nb*256+b][r*768+c] = x[b][(nb*2+r)*6+t][c]   (t<6, 768 rows, 1536 chans)
    long idx = (long)blockIdx.x * 256 + threadIdx.x;
    const long total = 6L * 768 * 1536;
    if (idx >= total) return;
    int j = (int)(idx % 1536);
    long nt = idx / 1536;
    int n = (int)(nt % 768);
    int t = (int)(nt / 768);
    int nb = n / 256, b = n % 256;
    int r = j / 768, c = j % 768;
    seq[idx] = f2bf(x[((long)b * 36 + (nb * 2 + r) * 6 + t) * 768 + c]);
}

// ---------- WMMA bf16 GEMM:  C[m][n] = bias[n] + sum_k A[m][k] * Bw[n][k] ----------
// A is split: k < splitK from A0 (lda=splitK), else from A1 (lda=K-splitK).
// Tiles: block = 128x128, BK=32, 256 threads = 8 waves, wave owns 32x64.
// Double-buffered LDS; global->LDS via async-to-LDS (ASYNCcnt) when available.
__global__ __launch_bounds__(256) void gemm_bf16_wmma(
    const ushort_t* __restrict__ A0, const ushort_t* __restrict__ A1, int splitK,
    const ushort_t* __restrict__ Bw, const float* __restrict__ bias,
    float* __restrict__ C, int M, int N, int K)
{
    __shared__ ushort_t As[2][128 * LDS_STRIDE];
    __shared__ ushort_t Bs[2][128 * LDS_STRIDE];

    const int tid  = threadIdx.x;
    const int lane = tid & 31;
    const int wave = tid >> 5;
    const int half = lane >> 4;     // 0 or 1
    const int fr   = lane & 15;

    const int bm = blockIdx.y * 128;
    const int bn = blockIdx.x * 128;
    const int wr = (wave >> 1) * 32;   // wave row offset within 128
    const int wc = (wave & 1) * 64;    // wave col offset within 128

    v8f acc[2][4];
#pragma unroll
    for (int i = 0; i < 2; i++)
#pragma unroll
        for (int j = 0; j < 4; j++)
#pragma unroll
            for (int e = 0; e < 8; e++) acc[i][j][e] = 0.0f;

    // global->LDS mapping: thread handles 16 contiguous bf16 (32B) of one tile row
    const int ldRow = tid >> 1;          // 0..127
    const int ldKp  = (tid & 1) * 16;    // 0 or 16
    const int ldsOff = ldRow * LDS_STRIDE + ldKp;

    // stage one 128x32 A tile + 128x32 B tile for k-chunk starting at k0 into buffer `buf`
    auto stage = [&](int buf, int k0) {
        int kk = k0 + ldKp;
        const ushort_t* sa = (kk < splitK)
            ? (A0 + (size_t)(bm + ldRow) * splitK + kk)
            : (A1 + (size_t)(bm + ldRow) * (K - splitK) + (kk - splitK));
        const ushort_t* sb = Bw + (size_t)(bn + ldRow) * K + kk;
#if HAS_ASYNC_LDS
        __builtin_amdgcn_global_load_async_to_lds_b128(
            (as1_v4i*)const_cast<ushort_t*>(sa), (as3_v4i*)&As[buf][ldsOff], 0, 0);
        __builtin_amdgcn_global_load_async_to_lds_b128(
            (as1_v4i*)const_cast<ushort_t*>(sa), (as3_v4i*)&As[buf][ldsOff + 8], 16, 0);
        __builtin_amdgcn_global_load_async_to_lds_b128(
            (as1_v4i*)const_cast<ushort_t*>(sb), (as3_v4i*)&Bs[buf][ldsOff], 0, 0);
        __builtin_amdgcn_global_load_async_to_lds_b128(
            (as1_v4i*)const_cast<ushort_t*>(sb), (as3_v4i*)&Bs[buf][ldsOff + 8], 16, 0);
#else
        uint4 a0v = *(const uint4*)(sa);
        uint4 a1v = *(const uint4*)(sa + 8);
        uint4 b0v = *(const uint4*)(sb);
        uint4 b1v = *(const uint4*)(sb + 8);
        *(uint4*)&As[buf][ldsOff]     = a0v;
        *(uint4*)&As[buf][ldsOff + 8] = a1v;
        *(uint4*)&Bs[buf][ldsOff]     = b0v;
        *(uint4*)&Bs[buf][ldsOff + 8] = b1v;
#endif
    };

    const int nCh = K >> 5;
    stage(0, 0);                         // prologue: chunk 0 -> buffer 0

    for (int ch = 0; ch < nCh; ch++) {
        const int buf = ch & 1;
        if (ch + 1 < nCh) {
            stage(buf ^ 1, (ch + 1) << 5);   // async-fill other buffer during this chunk's compute
            WAIT_ASYNC(4);                   // our 4 newest may remain in flight; chunk `ch` is done
        } else {
            WAIT_ASYNC(0);
        }
        __syncthreads();                     // all threads' fills of `buf` visible

        union U { v16bf v; uint4 q[2]; };
        U afr[2];
#pragma unroll
        for (int i = 0; i < 2; i++) {
            const ushort_t* ap = &As[buf][(wr + i * 16 + fr) * LDS_STRIDE + half * 8];
            afr[i].q[0] = *(const uint4*)(ap);        // K = base+0..7   (this half-group)
            afr[i].q[1] = *(const uint4*)(ap + 16);   // K = base+16..23
        }
        U bfr[4];
#pragma unroll
        for (int j = 0; j < 4; j++) {
            const ushort_t* bp = &Bs[buf][(wc + j * 16 + fr) * LDS_STRIDE + half * 8];
            bfr[j].q[0] = *(const uint4*)(bp);        // K = 8*half + 0..7
            bfr[j].q[1] = *(const uint4*)(bp + 16);   // K = 16 + 8*half + 0..7
        }
#pragma unroll
        for (int i = 0; i < 2; i++)
#pragma unroll
            for (int j = 0; j < 4; j++)
                acc[i][j] = __builtin_amdgcn_wmma_f32_16x16x32_bf16(
                    false, afr[i].v, false, bfr[j].v,
                    (short)0, acc[i][j], false, false);
        __syncthreads();                     // done reading `buf`; next stage may overwrite it
    }

    // ---- epilogue: lane l -> (M = r + 8*half, N = fr) within each 16x16 tile ----
#pragma unroll
    for (int i = 0; i < 2; i++)
#pragma unroll
        for (int j = 0; j < 4; j++) {
            int cm = bm + wr + i * 16 + half * 8;
            int cn = bn + wc + j * 16 + fr;
            float bb = bias ? bias[cn] : 0.0f;
#pragma unroll
            for (int r = 0; r < 8; r++)
                C[(size_t)(cm + r) * N + cn] = acc[i][j][r] + bb;
        }
}

// ---------- LSTM cell elementwise ----------
__global__ __launch_bounds__(256) void lstm_cell(
    const float* __restrict__ G /*768 x 6144, gates i,f,g,o*/,
    float* __restrict__ c /*768 x 1536*/,
    ushort_t* __restrict__ dA, ushort_t* __restrict__ dB,
    ushort_t* __restrict__ attScatter, int step)
{
    int idx = blockIdx.x * 256 + threadIdx.x;
    if (idx >= 768 * 1536) return;
    int m = idx / 1536, j = idx % 1536;
    const float* g = G + (size_t)m * 6144;
    float i_ = sigm(g[j]);
    float f_ = sigm(g[1536 + j]);
    float gg = tanhf(g[3072 + j]);
    float o_ = sigm(g[4608 + j]);
    float cn = f_ * c[idx] + i_ * gg;
    c[idx] = cn;
    float h = o_ * tanhf(cn);
    ushort_t hb = f2bf(h);
    dA[idx] = hb;
    if (dB) dB[idx] = hb;
    if (attScatter) {
        int r = j / 768, cc = j % 768;
        int a = m * 12 + r * 6 + step;   // row in (NB*B*T) attention layout
        attScatter[(size_t)a * 768 + cc] = hb;
    }
}

// ---------- attention core (per block, batch, head): softmax(QK^T/sqrt(dk)) V ----------
__global__ __launch_bounds__(256) void attn_core(
    const float* __restrict__ Q, const float* __restrict__ Kx, const float* __restrict__ V,
    ushort_t* __restrict__ O /*bf16, 9216 x 512*/)
{
    int bid = blockIdx.x;          // [0, 768*4)
    int h = bid & 3;
    int mb = bid >> 2;             // nb*256 + b
    int nb = mb / 256, b = mb % 256;
    int qn = (nb == 1) ? 0 : 1;    // qmap = [1,0,1]
    int kn = (nb == 2) ? 0 : 2;    // kmap = [2,2,0]
    int qrow0 = (qn * 256 + b) * 12;
    int krow0 = (kn * 256 + b) * 12;
    int vrow0 = mb * 12;

    __shared__ float Qs[12][128], Ks[12][128], Vs[12][128];
    __shared__ float Ss[12][13];

    int tid = threadIdx.x;
    for (int i = tid; i < 12 * 128; i += 256) {
        int t = i >> 7, d = i & 127;
        Qs[t][d] = Q [(size_t)(qrow0 + t) * 512 + h * 128 + d];
        Ks[t][d] = Kx[(size_t)(krow0 + t) * 512 + h * 128 + d];
        Vs[t][d] = V [(size_t)(vrow0 + t) * 512 + h * 128 + d];
    }
    __syncthreads();

    if (tid < 144) {
        int q = tid / 12, k = tid % 12;
        float s = 0.0f;
        for (int d = 0; d < 128; d++) s += Qs[q][d] * Ks[k][d];
        Ss[q][k] = s * 0.08838834764831845f;   // 1/sqrt(128)
    }
    __syncthreads();
    if (tid < 12) {
        float mx = Ss[tid][0];
        for (int k = 1; k < 12; k++) mx = fmaxf(mx, Ss[tid][k]);
        float sum = 0.0f;
        for (int k = 0; k < 12; k++) { float e = __expf(Ss[tid][k] - mx); Ss[tid][k] = e; sum += e; }
        float inv = 1.0f / sum;
        for (int k = 0; k < 12; k++) Ss[tid][k] *= inv;
    }
    __syncthreads();
    if (tid < 128) {
        for (int q = 0; q < 12; q++) {
            float o = 0.0f;
            for (int k = 0; k < 12; k++) o += Ss[q][k] * Vs[k][tid];
            O[(size_t)(vrow0 + q) * 512 + h * 128 + tid] = f2bf(o);
        }
    }
}

// ---------- LayerNorm + reorder + crop ----------
__global__ __launch_bounds__(256) void ln_reorder(
    const float* __restrict__ X /*9216 x 768*/,
    const float* __restrict__ gam, const float* __restrict__ bet,
    float* __restrict__ out)
{
    int a = blockIdx.x;            // 0..9215
    const float* row = X + (size_t)a * 768;
    __shared__ float red[256];
    int tid = threadIdx.x;

    float s = 0.0f;
    for (int c = tid; c < 768; c += 256) s += row[c];
    red[tid] = s; __syncthreads();
    for (int st = 128; st > 0; st >>= 1) { if (tid < st) red[tid] += red[tid + st]; __syncthreads(); }
    float mu = red[0] / 768.0f;
    __syncthreads();

    float v = 0.0f;
    for (int c = tid; c < 768; c += 256) { float d = row[c] - mu; v += d * d; }
    red[tid] = v; __syncthreads();
    for (int st = 128; st > 0; st >>= 1) { if (tid < st) red[tid] += red[tid + st]; __syncthreads(); }
    float inv = rsqrtf(red[0] / 768.0f + 1e-5f);

    int tt = a % 12, mb = a / 12;
    int r = tt / 6, w = tt % 6;
    int nb = mb / 256, b = mb % 256;
    int hh = nb * 2 + r;
    float* feat = out + ((size_t)b * 36 + hh * 6 + w) * 768;
    float* od = nullptr;
    if (hh >= 1 && hh <= 4 && w >= 1 && w <= 4)
        od = out + 7077888 + (((size_t)b * 4 + (hh - 1)) * 4 + (w - 1)) * 768;

    for (int c = tid; c < 768; c += 256) {
        float y = (row[c] - mu) * inv * gam[c] + bet[c];
        feat[c] = y;
        if (od) od[c] = y;
    }
}

// ---------- host ----------
extern "C" void kernel_launch(void* const* d_in, const int* in_sizes, int n_in,
                              void* d_out, int out_size, void* d_ws, size_t ws_size,
                              hipStream_t stream) {
    const float* x       = (const float*)d_in[0];
    const float* enc_Wih = (const float*)d_in[1];
    const float* enc_Whh = (const float*)d_in[2];
    const float* enc_bih = (const float*)d_in[3];
    const float* enc_bhh = (const float*)d_in[4];
    const float* dec_Wih = (const float*)d_in[5];
    const float* dec_Whh = (const float*)d_in[6];
    const float* dec_bih = (const float*)d_in[7];
    const float* dec_bhh = (const float*)d_in[8];
    const float* Wq = (const float*)d_in[9];  const float* bq = (const float*)d_in[10];
    const float* Wk = (const float*)d_in[11]; const float* bk = (const float*)d_in[12];
    const float* Wv = (const float*)d_in[13]; const float* bv = (const float*)d_in[14];
    const float* Wo = (const float*)d_in[15]; const float* bo = (const float*)d_in[16];
    const float* ln_g = (const float*)d_in[17];
    const float* ln_b = (const float*)d_in[18];

    // ---- workspace layout ----
    char* ws = (char*)d_ws;
    size_t off = 0;
    auto alloc = [&](size_t bytes) { size_t o = off; off += (bytes + 255) & ~(size_t)255; return o; };
    const size_t WLSTM = 2ull * 6144 * 3072 * 2;          // bf16
    ushort_t* wenc = (ushort_t*)(ws + alloc(WLSTM));
    ushort_t* wdec = (ushort_t*)(ws + alloc(WLSTM));
    ushort_t* wqT  = (ushort_t*)(ws + alloc(512ull * 768 * 2));
    ushort_t* wkT  = (ushort_t*)(ws + alloc(512ull * 768 * 2));
    ushort_t* wvT  = (ushort_t*)(ws + alloc(512ull * 768 * 2));
    ushort_t* woT  = (ushort_t*)(ws + alloc(768ull * 512 * 2));
    float*    benc = (float*)(ws + alloc(2ull * 6144 * 4));
    float*    bdec = (float*)(ws + alloc(2ull * 6144 * 4));
    ushort_t* seqb = (ushort_t*)(ws + alloc(6ull * 768 * 1536 * 2));
    ushort_t* h0   = (ushort_t*)(ws + alloc(768ull * 1536 * 2));
    ushort_t* h1   = (ushort_t*)(ws + alloc(768ull * 1536 * 2));
    ushort_t* yb   = (ushort_t*)(ws + alloc(768ull * 1536 * 2));
    float*    c0   = (float*)(ws + alloc(768ull * 1536 * 4));
    float*    c1   = (float*)(ws + alloc(768ull * 1536 * 4));
    float*    G    = (float*)(ws + alloc(768ull * 6144 * 4));
    ushort_t* attin= (ushort_t*)(ws + alloc(9216ull * 768 * 2));
    float*    Qall = (float*)(ws + alloc(9216ull * 512 * 4));
    float*    Kall = (float*)(ws + alloc(9216ull * 512 * 4));
    float*    Vall = (float*)(ws + alloc(9216ull * 512 * 4));
    ushort_t* attO = (ushort_t*)(ws + alloc(9216ull * 512 * 2));
    float*    O2   = (float*)(ws + alloc(9216ull * 768 * 4));
    (void)ws_size; (void)in_sizes; (void)n_in; (void)out_size;

    // ---- pack weights / inputs ----
    {
        long tot = 2L * 6144 * 3072;
        int grid = (int)((tot + 255) / 256);
        pack_lstm_w<<<grid, 256, 0, stream>>>(enc_Wih, enc_Whh, wenc);
        pack_lstm_w<<<grid, 256, 0, stream>>>(dec_Wih, dec_Whh, wdec);
    }
    pack_bias<<<(2 * 6144 + 255) / 256, 256, 0, stream>>>(enc_bih, enc_bhh, benc, 2 * 6144);
    pack_bias<<<(2 * 6144 + 255) / 256, 256, 0, stream>>>(dec_bih, dec_bhh, bdec, 2 * 6144);
    transpose_w<<<(512 * 768 + 255) / 256, 256, 0, stream>>>(Wq, wqT, 768, 512);
    transpose_w<<<(512 * 768 + 255) / 256, 256, 0, stream>>>(Wk, wkT, 768, 512);
    transpose_w<<<(512 * 768 + 255) / 256, 256, 0, stream>>>(Wv, wvT, 768, 512);
    transpose_w<<<(768 * 512 + 255) / 256, 256, 0, stream>>>(Wo, woT, 512, 768);
    build_seq<<<(int)((6L * 768 * 1536 + 255) / 256), 256, 0, stream>>>(x, seqb);

    (void)hipMemsetAsync(h0, 0, 768ull * 1536 * 2, stream);
    (void)hipMemsetAsync(h1, 0, 768ull * 1536 * 2, stream);
    (void)hipMemsetAsync(c0, 0, 768ull * 1536 * 4, stream);
    (void)hipMemsetAsync(c1, 0, 768ull * 1536 * 4, stream);

    const dim3 gatesGrid(6144 / 128, 768 / 128);
    const int cellGrid = (768 * 1536) / 256;

    // ---- encoder: consumes columns reversed (t = 5..0) ----
    for (int s = 0; s < 6; s++) {
        int t = 5 - s;
        const ushort_t* xt = seqb + (size_t)t * 768 * 1536;
        // layer 0
        gemm_bf16_wmma<<<gatesGrid, 256, 0, stream>>>(xt, h0, 1536, wenc, benc, G, 768, 6144, 3072);
        lstm_cell<<<cellGrid, 256, 0, stream>>>(G, c0, h0, nullptr, nullptr, 0);
        // layer 1
        gemm_bf16_wmma<<<gatesGrid, 256, 0, stream>>>(h0, h1, 1536,
            wenc + 6144ull * 3072, benc + 6144, G, 768, 6144, 3072);
        lstm_cell<<<cellGrid, 256, 0, stream>>>(G, c1, h1, nullptr, nullptr, 0);
    }
    // decoder first input = en_out = final h1
    (void)hipMemcpyAsync(yb, h1, 768ull * 1536 * 2, hipMemcpyDeviceToDevice, stream);

    // ---- decoder: autoregressive, 6 steps ----
    for (int t = 0; t < 6; t++) {
        gemm_bf16_wmma<<<gatesGrid, 256, 0, stream>>>(yb, h0, 1536, wdec, bdec, G, 768, 6144, 3072);
        lstm_cell<<<cellGrid, 256, 0, stream>>>(G, c0, h0, nullptr, nullptr, 0);
        gemm_bf16_wmma<<<gatesGrid, 256, 0, stream>>>(h0, h1, 1536,
            wdec + 6144ull * 3072, bdec + 6144, G, 768, 6144, 3072);
        lstm_cell<<<cellGrid, 256, 0, stream>>>(G, c1, h1, yb, attin, t);
    }

    // ---- attention projections (M=9216, K=768, N=512) ----
    gemm_bf16_wmma<<<dim3(512 / 128, 9216 / 128), 256, 0, stream>>>(attin, attin, 768, wqT, bq, Qall, 9216, 512, 768);
    gemm_bf16_wmma<<<dim3(512 / 128, 9216 / 128), 256, 0, stream>>>(attin, attin, 768, wkT, bk, Kall, 9216, 512, 768);
    gemm_bf16_wmma<<<dim3(512 / 128, 9216 / 128), 256, 0, stream>>>(attin, attin, 768, wvT, bv, Vall, 9216, 512, 768);

    attn_core<<<768 * 4, 256, 0, stream>>>(Qall, Kall, Vall, attO);

    // ---- output projection (M=9216, K=512, N=768) ----
    gemm_bf16_wmma<<<dim3(768 / 128, 9216 / 128), 256, 0, stream>>>(attO, attO, 512, woT, bo, O2, 9216, 768, 512);

    // ---- layernorm + reorder + crop into d_out (feature, out_decode) ----
    ln_reorder<<<9216, 256, 0, stream>>>(O2, ln_g, ln_b, (float*)d_out);
}
